// CustomSpecificity_78907139162812
// MI455X (gfx1250) — matrix operand order repaired and checked by reading, not compiled, not run on hardware
//
#include <hip/hip_runtime.h>

#define NUM_CLASSES 16
#define CM_ELEMS (NUM_CLASSES * NUM_CLASSES)
#define EPS_F32 1.1920928955078125e-07f

typedef __attribute__((ext_vector_type(16))) _Float16 v16h;
typedef __attribute__((ext_vector_type(8)))  float    v8f;

// ---------------------------------------------------------------------------
// Kernel 0: zero the 16x16 confusion-matrix accumulator in workspace.
// ---------------------------------------------------------------------------
__global__ void spec_zero_cm(float* __restrict__ cm) {
    const int t = threadIdx.x;
    if (t < CM_ELEMS) cm[t] = 0.0f;
}

// ---------------------------------------------------------------------------
// Kernel 1: per-row argmax + confusion matrix via one-hot WMMA.
//   cm = OneHot(t)^T * OneHot(p), accumulated with v_wmma_f32_16x16x32_f16.
//   Each wave handles 32 rows per WMMA (K=32). Counts are exact in f32.
// ---------------------------------------------------------------------------
__global__ __launch_bounds__(256) void spec_confusion_wmma(
    const float* __restrict__ y_true,
    const float* __restrict__ y_pred,
    float* __restrict__ cm_global,
    int N)
{
    __shared__ float cm_lds[CM_ELEMS];

    const int tid           = threadIdx.x;
    const int lane          = tid & 31;
    const int waveInBlock   = tid >> 5;
    const int wavesPerBlock = blockDim.x >> 5;
    const int globalWave    = blockIdx.x * wavesPerBlock + waveInBlock;
    const int totalWaves    = gridDim.x * wavesPerBlock;

    if (tid < CM_ELEMS) cm_lds[tid] = 0.0f;
    __syncthreads();

    const int nChunks = (N + 31) >> 5;
    const int hsel    = (lane >= 16) ? 1 : 0;   // which half-wave
    const int m       = lane & 15;              // M (A row) == N (B col) for this lane

    v8f c = {};  // 16x16 f32 accumulator fragment (partial confusion matrix)

    for (int chunk = globalWave; chunk < nChunks; chunk += totalWaves) {
        const long row = (long)chunk * 32 + lane;

        // --- per-lane argmax over 16 classes (first-max, matches jnp.argmax) ---
        int tIdx = -1, pIdx = -1;   // -1 => inactive row, contributes 0 one-hot
        if (row < (long)N) {
            const float4* rt = (const float4*)(y_true + row * NUM_CLASSES);
            const float4* rp = (const float4*)(y_pred + row * NUM_CLASSES);
            float4 t0 = rt[0], t1 = rt[1], t2 = rt[2], t3 = rt[3];
            float4 p0 = rp[0], p1 = rp[1], p2 = rp[2], p3 = rp[3];

            float tv[16] = {t0.x,t0.y,t0.z,t0.w, t1.x,t1.y,t1.z,t1.w,
                            t2.x,t2.y,t2.z,t2.w, t3.x,t3.y,t3.z,t3.w};
            float pv[16] = {p0.x,p0.y,p0.z,p0.w, p1.x,p1.y,p1.z,p1.w,
                            p2.x,p2.y,p2.z,p2.w, p3.x,p3.y,p3.z,p3.w};

            float tb = tv[0], pb = pv[0];
            tIdx = 0; pIdx = 0;
#pragma unroll
            for (int i = 1; i < 16; ++i) {
                if (tv[i] > tb) { tb = tv[i]; tIdx = i; }
                if (pv[i] > pb) { pb = pv[i]; pIdx = i; }
            }
        }

        // --- build one-hot WMMA fragments via wave32 bpermute shuffles ---
        // A (16x32 f16, MxK): lane holds row M=lane%16; half index h maps to
        //   VGPR i=h/2, sub=h%2:  lanes 0-15: i<4 -> K=2i+sub,  i>=4 -> K=16+2(i-4)+sub
        //                         lanes16-31: i<4 -> K=8+2i+sub, i>=4 -> K=24+2(i-4)+sub
        // B (32x16 f16, KxN): lane holds col N=lane%16; half h -> same K as A
        v16h af, bf;
#pragma unroll
        for (int h = 0; h < 16; ++h) {
            const int i   = h >> 1;
            const int sub = h & 1;
            const int kA  = (i < 4) ? ((hsel ? 8 : 0) + 2 * i + sub)
                                    : ((hsel ? 24 : 16) + 2 * (i - 4) + sub);
            const int kB  = kA;
            const int tk  = __shfl(tIdx, kA, 32);
            const int pk  = __shfl(pIdx, kB, 32);
            af[h] = (tk == m) ? (_Float16)1.0f : (_Float16)0.0f;
            bf[h] = (pk == m) ? (_Float16)1.0f : (_Float16)0.0f;
        }

        // D = A*B + C : accumulates 32 (t,p) pairs into the 16x16 count matrix
        c = __builtin_amdgcn_wmma_f32_16x16x32_f16(
                /*neg_a=*/false, af, /*neg_b=*/false, bf,
                /*c_mod=*/(short)0, c, /*reuse_a=*/false, /*reuse_b=*/false);
    }

    // --- reduce wave fragments: C/D layout: VGPR r, lanes 0-15 -> M=r,
    //     lanes 16-31 -> M=r+8; column N = lane%16. ---
#pragma unroll
    for (int r = 0; r < 8; ++r) {
        const int rowM = r + (hsel ? 8 : 0);
        atomicAdd(&cm_lds[rowM * NUM_CLASSES + m], c[r]);
    }
    __syncthreads();

    if (tid < CM_ELEMS) {
        const float v = cm_lds[tid];
        if (v != 0.0f) atomicAdd(&cm_global[tid], v);
    }
}

// ---------------------------------------------------------------------------
// Kernel 2: specificity from confusion matrix (macro average).
// ---------------------------------------------------------------------------
__global__ void spec_finalize(const float* __restrict__ cm, float* __restrict__ out) {
    __shared__ float spec[NUM_CLASSES];
    const int tid = threadIdx.x;
    if (tid < NUM_CLASSES) {
        float rowsum = 0.0f, colsum = 0.0f, total = 0.0f;
#pragma unroll
        for (int j = 0; j < NUM_CLASSES; ++j) {
            rowsum += cm[tid * NUM_CLASSES + j];
            colsum += cm[j * NUM_CLASSES + tid];
        }
        for (int k = 0; k < CM_ELEMS; ++k) total += cm[k];
        const float tp = cm[tid * NUM_CLASSES + tid];
        const float fp = colsum - tp;
        const float fn = rowsum - tp;
        const float tn = total - (tp + fp + fn);
        spec[tid] = tn / (tn + fp + EPS_F32);
    }
    __syncthreads();
    if (tid == 0) {
        float s = 0.0f;
#pragma unroll
        for (int cix = 0; cix < NUM_CLASSES; ++cix) s += spec[cix];
        out[0] = s / (float)NUM_CLASSES;
    }
}

// ---------------------------------------------------------------------------
extern "C" void kernel_launch(void* const* d_in, const int* in_sizes, int n_in,
                              void* d_out, int out_size, void* d_ws, size_t ws_size,
                              hipStream_t stream) {
    const float* y_true = (const float*)d_in[0];
    const float* y_pred = (const float*)d_in[1];
    float* out = (float*)d_out;
    float* cm  = (float*)d_ws;           // 256 floats of scratch

    const int N = in_sizes[0] / NUM_CLASSES;   // 2,000,000

    spec_zero_cm<<<1, 256, 0, stream>>>(cm);

    // 1024 blocks x 8 wave32 = 8192 waves; ~8 chunks (32 rows, 4 KB) per wave.
    // 256 MB total traffic -> ~11 us at 23.3 TB/s HBM roofline.
    spec_confusion_wmma<<<1024, 256, 0, stream>>>(y_true, y_pred, cm, N);

    spec_finalize<<<1, 256, 0, stream>>>(cm, out);
}
